// SwinTransformerBlock_32469952757776
// MI455X (gfx1250) — matrix operand order repaired and checked by reading, not compiled, not run on hardware
//
#include <hip/hip_runtime.h>
#include <hip/hip_bf16.h>
#include <stdint.h>

// ---------------------------------------------------------------------------
// Swin Transformer block on MI455X (gfx1250, wave32, WMMA 16x16x32 f16)
// B=64 H=W=28 C=384 NH=12 HD=32 WS=7 SHIFT=3 HID=1536
// tokens = 50176, windows = 1024 (padded 49 -> 64 rows)
// ---------------------------------------------------------------------------

typedef __attribute__((ext_vector_type(16))) _Float16 v16h;
typedef __attribute__((ext_vector_type(8)))  float    v8f;
typedef __attribute__((ext_vector_type(4)))  float    f32x4;

#define SCALE_F 0.17677669529663687f   // 32^-0.5

union AF { v16h v; uint32_t u[8]; f32x4 f4[2]; };

// LDS offset of a flat pointer to __shared__ (aperture: low 32 bits = LDS addr)
__device__ __forceinline__ uint32_t lds_addr(const void* p) {
    return (uint32_t)(uintptr_t)p;
}

// ---------------------------------------------------------------------------
// f32 -> f16 weight conversion
// ---------------------------------------------------------------------------
__global__ void cvt_f32_f16(const float* __restrict__ in, _Float16* __restrict__ out, int n) {
    int i = blockIdx.x * blockDim.x + threadIdx.x;
    if (i < n) out[i] = (_Float16)in[i];
}

// Zero pad rows 49..63 of each window tile of xw  (1024 * 15 * 384 halfs)
__global__ void zero_pad_rows(_Float16* __restrict__ xw) {
    int idx = blockIdx.x * blockDim.x + threadIdx.x;   // each thread: 8 halfs
    int e   = idx * 8;
    int win = e / 5760;            // 15*384
    int rem = e % 5760;
    f32x4 z = {};
    *(f32x4*)&xw[(size_t)win * 24576 + 49 * 384 + rem] = z;
}

// ---------------------------------------------------------------------------
// LayerNorm 1 fused with cyclic shift (-3,-3) and window partition -> f16
// one wave (32 lanes) per token; 8 tokens per 256-thread block
// ---------------------------------------------------------------------------
__global__ __launch_bounds__(256)
void ln1_shift_window(const float* __restrict__ x, const float* __restrict__ g,
                      const float* __restrict__ bta, _Float16* __restrict__ xw) {
    int wave = threadIdx.x >> 5, lane = threadIdx.x & 31;
    int tok = blockIdx.x * 8 + wave;                   // < 50176
    const float* xp = x + (size_t)tok * 384;
    float vals[12];
    float s = 0.f, s2 = 0.f;
#pragma unroll
    for (int i = 0; i < 12; ++i) {
        float v = xp[lane + 32 * i];
        vals[i] = v; s += v; s2 += v * v;
    }
#pragma unroll
    for (int o = 16; o > 0; o >>= 1) { s += __shfl_xor(s, o, 32); s2 += __shfl_xor(s2, o, 32); }
    float mu  = s  * (1.f / 384.f);
    float var = s2 * (1.f / 384.f) - mu * mu;
    float inv = rsqrtf(var + 1e-5f);

    int b = tok / 784, rem = tok % 784;
    int hh = rem / 28, ww = rem % 28;
    int si = hh - 3; if (si < 0) si += 28;             // roll(-3)
    int sj = ww - 3; if (sj < 0) sj += 28;
    int win = b * 16 + (si / 7) * 4 + (sj / 7);
    int tw  = (si % 7) * 7 + (sj % 7);
    _Float16* dp = xw + ((size_t)win * 64 + tw) * 384;
#pragma unroll
    for (int i = 0; i < 12; ++i) {
        int ch = lane + 32 * i;
        dp[ch] = (_Float16)((vals[i] - mu) * inv * g[ch] + bta[ch]);
    }
}

// LayerNorm 2 -> f16 rows [50176, 384]
__global__ __launch_bounds__(256)
void ln2_to_f16(const float* __restrict__ x, const float* __restrict__ g,
                const float* __restrict__ bta, _Float16* __restrict__ out) {
    int wave = threadIdx.x >> 5, lane = threadIdx.x & 31;
    int tok = blockIdx.x * 8 + wave;
    const float* xp = x + (size_t)tok * 384;
    float vals[12];
    float s = 0.f, s2 = 0.f;
#pragma unroll
    for (int i = 0; i < 12; ++i) {
        float v = xp[lane + 32 * i];
        vals[i] = v; s += v; s2 += v * v;
    }
#pragma unroll
    for (int o = 16; o > 0; o >>= 1) { s += __shfl_xor(s, o, 32); s2 += __shfl_xor(s2, o, 32); }
    float mu  = s  * (1.f / 384.f);
    float var = s2 * (1.f / 384.f) - mu * mu;
    float inv = rsqrtf(var + 1e-5f);
    _Float16* dp = out + (size_t)tok * 384;
#pragma unroll
    for (int i = 0; i < 12; ++i) {
        int ch = lane + 32 * i;
        dp[ch] = (_Float16)((vals[i] - mu) * inv * g[ch] + bta[ch]);
    }
}

// ---------------------------------------------------------------------------
// GEMM epilogues
// ---------------------------------------------------------------------------
struct EpiBiasHalf {                    // qkv: out_f16 = v + bias
    _Float16* out; const float* bias; int N;
    __device__ void operator()(int row, int col, float v) const {
        out[(size_t)row * N + col] = (_Float16)(v + bias[col]);
    }
};
struct EpiGeluHalf {                    // fc1: out_f16 = gelu(v + bias), exact gelu
    _Float16* out; const float* bias; int N;
    __device__ void operator()(int row, int col, float v) const {
        float xv = v + bias[col];
        float gl = 0.5f * xv * (1.f + erff(xv * 0.70710678118654752f));
        out[(size_t)row * N + col] = (_Float16)gl;
    }
};
struct EpiProjScatter {                 // proj: window-reverse + un-shift + residual
    float* xmid; const float* xin; const float* bias;
    __device__ void operator()(int row, int col, float v) const {
        int tw = row & 63;
        if (tw >= 49) return;           // drop pad rows
        int win = row >> 6;
        int b = win >> 4, nw = win & 15;
        int wh = nw >> 2, wwn = nw & 3;
        int r = tw / 7, c = tw % 7;
        int sh = wh * 7 + r, sw = wwn * 7 + c;
        int oh = sh + 3; if (oh >= 28) oh -= 28;       // roll(+3)
        int ow = sw + 3; if (ow >= 28) ow -= 28;
        size_t idx = ((size_t)b * 784 + oh * 28 + ow) * 384 + col;
        xmid[idx] = xin[idx] + v + bias[col];
    }
};
struct EpiFc2Out {                      // fc2: out = xmid + v + bias
    float* out; const float* xmid; const float* bias; int N;
    __device__ void operator()(int row, int col, float v) const {
        size_t idx = (size_t)row * N + col;
        out[idx] = xmid[idx] + v + bias[col];
    }
};

// ---------------------------------------------------------------------------
// Generic WMMA GEMM: C[M,N] = A[M,K] * B[K,N]   (f16 in, f32 acc)
// 256 threads = 8 waves. Workgroup tile: 128(M) x 64(N), K step 32.
// A tile staged with GLOBAL_LOAD_ASYNC_TO_LDS_B128 (ASYNCcnt),
// B tile transposed to N-major via scalar gather, so both WMMA fragments
// are contiguous dword/b128 LDS loads with the same contiguous k = hl*16 + e order in A and B.
// ---------------------------------------------------------------------------
template <typename Epi>
__global__ __launch_bounds__(256)
void wmma_gemm(const _Float16* __restrict__ A, const _Float16* __restrict__ Bm,
               int K, int N, Epi epi) {
    __shared__ alignas(16) _Float16 As[128 * 48];
    __shared__ alignas(16) _Float16 Bs[64 * 48];
    const int t = threadIdx.x;
    const int nBase = blockIdx.x * 64;
    const int mBase = blockIdx.y * 128;
    const int wave = t >> 5, lane = t & 31, hl = lane >> 4, ln = lane & 15;

    v8f acc[4] = {};

    for (int kk = 0; kk < K; kk += 32) {
        {   // async-DMA A tile 128x32 into LDS (two b128 per thread)
            int m = t >> 1, seg = (t & 1) * 16;
            uint32_t goff = (uint32_t)((((size_t)(mBase + m)) * K + kk + seg) * 2);
            uint32_t l0   = lds_addr(&As[m * 48 + seg]);
            asm volatile(
                "global_load_async_to_lds_b128 %0, %1, %2\n\t"
                "global_load_async_to_lds_b128 %3, %4, %2"
                :
                : "v"(l0), "v"(goff), "s"(A), "v"(l0 + 16u), "v"(goff + 16u)
                : "memory");
        }
        {   // stage B tile 32x64 transposed -> Bs[n][k]
            int n = t & 63, kr = (t >> 6) * 8;
            const _Float16* gp = Bm + (size_t)(kk + kr) * N + nBase + n;
#pragma unroll
            for (int i = 0; i < 8; ++i) Bs[n * 48 + kr + i] = gp[(size_t)i * N];
            if (kk + 32 < K) __builtin_prefetch(gp + (size_t)32 * N, 0, 0);
        }
        asm volatile("s_wait_asynccnt 0" ::: "memory");
        __syncthreads();

        AF a;
        {   // A fragment: lane = M row, k = hl*16 + e contiguous
            int m = wave * 16 + ln;
            a.f4[0] = *(const f32x4*)&As[m * 48 + hl * 16];
            a.f4[1] = *(const f32x4*)&As[m * 48 + hl * 16 + 8];
        }
#pragma unroll
        for (int nt = 0; nt < 4; ++nt) {
            AF bfr;
            int n = nt * 16 + ln;       // lane = N col, k = hl*16 + e contiguous
            bfr.f4[0] = *(const f32x4*)&Bs[n * 48 + hl * 16];
            bfr.f4[1] = *(const f32x4*)&Bs[n * 48 + hl * 16 + 8];
            acc[nt] = __builtin_amdgcn_wmma_f32_16x16x32_f16(
                false, a.v, false, bfr.v, (short)0, acc[nt], false, false);
        }
        __syncthreads();
    }
#pragma unroll
    for (int nt = 0; nt < 4; ++nt)
#pragma unroll
        for (int r = 0; r < 8; ++r)
            epi(mBase + wave * 16 + hl * 8 + r, nBase + nt * 16 + ln, acc[nt][r]);
}

// ---------------------------------------------------------------------------
// Windowed attention, one block per (head, window).
// S = (Q*scale) K^T  (64x64 padded, one WMMA per 16x16 tile, K=32)
// softmax with analytic rel-pos bias + shift mask, then O = P V.
// ---------------------------------------------------------------------------
__device__ __forceinline__ int regid(int g) { return g < 21 ? 0 : (g < 25 ? 1 : 2); }

__global__ __launch_bounds__(256)
void attn_kernel(const _Float16* __restrict__ qkv, const float* __restrict__ rpb,
                 _Float16* __restrict__ o) {
    __shared__ alignas(16) _Float16 Qs[64 * 48];    // A layout (k contiguous)
    __shared__ alignas(16) _Float16 Ks[64 * 48];    // B layout (n=token, k=hd contiguous)
    __shared__ alignas(16) _Float16 Vst[32 * 72];   // V transposed: [hd][token]
    __shared__ alignas(16) float    Sf[64 * 68];
    __shared__ alignas(16) _Float16 Ps[64 * 72];    // P as A for second GEMM

    const int head = blockIdx.x, win = blockIdx.y;
    const int t = threadIdx.x;
    const size_t base = (size_t)win * 64 * 1152 + head * 32;

    {   // K tile via async LDS DMA; Q tile via VALU (needs scaling)
        int m = t >> 2, off = (t & 3) * 8;
        uint32_t goff = (uint32_t)((base + (size_t)m * 1152 + 384 + off) * 2);
        uint32_t lk   = lds_addr(&Ks[m * 48 + off]);
        asm volatile("global_load_async_to_lds_b128 %0, %1, %2"
                     : : "v"(lk), "v"(goff), "s"(qkv) : "memory");

        const _Float16* gp = qkv + base + (size_t)m * 1152 + off;
        union { f32x4 f; _Float16 h[8]; } u;
        u.f = *(const f32x4*)gp;
#pragma unroll
        for (int i = 0; i < 8; ++i) u.h[i] = (_Float16)((float)u.h[i] * SCALE_F);
        *(f32x4*)&Qs[m * 48 + off] = u.f;
    }
    {   // load V transposed (scalar gather)
        int n = t & 31, kr = (t >> 5) * 8;
#pragma unroll
        for (int i = 0; i < 8; ++i)
            Vst[n * 72 + kr + i] = qkv[base + (size_t)(kr + i) * 1152 + 768 + n];
    }
    asm volatile("s_wait_asynccnt 0" ::: "memory");
    __syncthreads();

    const int wave = t >> 5, lane = t & 31, hl = lane >> 4, ln = lane & 15;

    // ---- S = Q K^T : 16 tiles; wave handles tiles (wave, wave+8), unrolled
#pragma unroll
    for (int it = 0; it < 2; ++it) {
        int tile = wave + it * 8;
        int mt = tile >> 2, nt = tile & 3;
        AF a, b;
        int m = mt * 16 + ln;
        a.f4[0] = *(const f32x4*)&Qs[m * 48 + hl * 16];
        a.f4[1] = *(const f32x4*)&Qs[m * 48 + hl * 16 + 8];
        int n = nt * 16 + ln;
        b.f4[0] = *(const f32x4*)&Ks[n * 48 + hl * 16];
        b.f4[1] = *(const f32x4*)&Ks[n * 48 + hl * 16 + 8];
        v8f acc = {};
        acc = __builtin_amdgcn_wmma_f32_16x16x32_f16(false, a.v, false, b.v,
                                                     (short)0, acc, false, false);
#pragma unroll
        for (int r = 0; r < 8; ++r)
            Sf[(mt * 16 + hl * 8 + r) * 68 + nt * 16 + ln] = acc[r];
    }
    __syncthreads();

    // ---- bias + mask + softmax: 4 threads per row, 16 cols each,
    //      row max/sum via shfl_xor over the 4-lane group (wave32)
    {
        int i = t >> 2, q = t & 3, j0 = q * 16;
        int nw = win & 15, wh = nw >> 2, wwn = nw & 3;
        bool vr = i < 49;
        int ih = 0, iw = 0, idi = 0;
        if (vr) { ih = i / 7; iw = i - ih * 7;
                  idi = regid(wh * 7 + ih) * 3 + regid(wwn * 7 + iw); }
        float vals[16];
        float mx = -1e30f;
#pragma unroll
        for (int jj = 0; jj < 16; ++jj) {
            int j = j0 + jj;
            float val;
            if (!vr)          val = 0.f;      // pad row: uniform (discarded later)
            else if (j >= 49) val = -1e9f;    // pad col: masked out
            else {
                int jh = j / 7, jw = j - jh * 7;
                int rel = (ih - jh + 6) * 13 + (iw - jw + 6);
                int idj = regid(wh * 7 + jh) * 3 + regid(wwn * 7 + jw);
                val = Sf[i * 68 + j] + rpb[rel * 12 + head]
                    + (idi == idj ? 0.f : -100.f);
            }
            vals[jj] = val;
            mx = fmaxf(mx, val);
        }
        mx = fmaxf(mx, __shfl_xor(mx, 1, 32));
        mx = fmaxf(mx, __shfl_xor(mx, 2, 32));
        float sum = 0.f;
#pragma unroll
        for (int jj = 0; jj < 16; ++jj) {
            float e = __expf(vals[jj] - mx);
            vals[jj] = e; sum += e;
        }
        sum += __shfl_xor(sum, 1, 32);
        sum += __shfl_xor(sum, 2, 32);
        float inv = 1.f / sum;
#pragma unroll
        for (int jj = 0; jj < 16; ++jj)
            Ps[i * 72 + j0 + jj] = (_Float16)(vals[jj] * inv);
    }
    __syncthreads();

    // ---- O = P V : 8 tiles (4m x 2n), one per wave, K = 64 (two steps)
    {
        int mt = wave >> 1, nt = wave & 1;
        v8f acc = {};
#pragma unroll
        for (int k0 = 0; k0 < 64; k0 += 32) {
            AF a, b;
            int m = mt * 16 + ln;
            a.f4[0] = *(const f32x4*)&Ps[m * 72 + k0 + hl * 16];
            a.f4[1] = *(const f32x4*)&Ps[m * 72 + k0 + hl * 16 + 8];
            int n = nt * 16 + ln;
            b.f4[0] = *(const f32x4*)&Vst[n * 72 + k0 + hl * 16];
            b.f4[1] = *(const f32x4*)&Vst[n * 72 + k0 + hl * 16 + 8];
            acc = __builtin_amdgcn_wmma_f32_16x16x32_f16(false, a.v, false, b.v,
                                                         (short)0, acc, false, false);
        }
#pragma unroll
        for (int r = 0; r < 8; ++r) {
            int row = win * 64 + mt * 16 + hl * 8 + r;
            int col = head * 32 + nt * 16 + ln;
            o[(size_t)row * 384 + col] = (_Float16)acc[r];
        }
    }
}

// ---------------------------------------------------------------------------
// Host launcher
// ---------------------------------------------------------------------------
extern "C" void kernel_launch(void* const* d_in, const int* in_sizes, int n_in,
                              void* d_out, int out_size, void* d_ws, size_t ws_size,
                              hipStream_t stream) {
    (void)in_sizes; (void)n_in; (void)out_size; (void)ws_size;
    const float* x      = (const float*)d_in[0];
    const float* qkv_w  = (const float*)d_in[1];
    const float* qkv_b  = (const float*)d_in[2];
    const float* proj_w = (const float*)d_in[3];
    const float* proj_b = (const float*)d_in[4];
    const float* rpb    = (const float*)d_in[5];
    const float* n1g    = (const float*)d_in[6];
    const float* n1b    = (const float*)d_in[7];
    const float* n2g    = (const float*)d_in[8];
    const float* n2b    = (const float*)d_in[9];
    const float* fc1_w  = (const float*)d_in[10];
    const float* fc1_b  = (const float*)d_in[11];
    const float* fc2_w  = (const float*)d_in[12];
    const float* fc2_b  = (const float*)d_in[13];
    float* out = (float*)d_out;

    char* ws = (char*)d_ws;
    size_t off = 0;
    auto alloc = [&](size_t bytes) { size_t o = off; off += (bytes + 255) & ~(size_t)255; return o; };

    _Float16* wqkv  = (_Float16*)(ws + alloc(442368 * 2));     // 384x1152
    _Float16* wproj = (_Float16*)(ws + alloc(147456 * 2));     // 384x384
    _Float16* wfc1  = (_Float16*)(ws + alloc(589824 * 2));     // 384x1536
    _Float16* wfc2  = (_Float16*)(ws + alloc(589824 * 2));     // 1536x384
    size_t regA = alloc(201326592);   // phase1: xw(50.3MB)+qkv(151MB); phase2: h3(154MB)
    size_t regB = alloc(50331648);    // phase1: o_pad; phase2: h2
    size_t regC = alloc(77070336);    // x_mid (f32, attention residual output)

    _Float16* xw   = (_Float16*)(ws + regA);
    _Float16* qkvb = (_Float16*)(ws + regA + 50331648);
    _Float16* h3   = (_Float16*)(ws + regA);
    _Float16* opad = (_Float16*)(ws + regB);
    _Float16* h2   = (_Float16*)(ws + regB);
    float*    xmid = (float*)(ws + regC);

    // weights -> f16
    cvt_f32_f16<<<(442368 + 255) / 256, 256, 0, stream>>>(qkv_w, wqkv, 442368);
    cvt_f32_f16<<<(147456 + 255) / 256, 256, 0, stream>>>(proj_w, wproj, 147456);
    cvt_f32_f16<<<(589824 + 255) / 256, 256, 0, stream>>>(fc1_w, wfc1, 589824);
    cvt_f32_f16<<<(589824 + 255) / 256, 256, 0, stream>>>(fc2_w, wfc2, 589824);

    // LN1 + shift + window partition (padded 49->64)
    zero_pad_rows<<<2880, 256, 0, stream>>>(xw);
    ln1_shift_window<<<6272, 256, 0, stream>>>(x, n1g, n1b, xw);

    // QKV GEMM: [65536,384] x [384,1152]
    wmma_gemm<EpiBiasHalf><<<dim3(18, 512), 256, 0, stream>>>(
        xw, wqkv, 384, 1152, EpiBiasHalf{qkvb, qkv_b, 1152});

    // attention per (head, window)
    attn_kernel<<<dim3(12, 1024), 256, 0, stream>>>(qkvb, rpb, opad);

    // proj GEMM + window reverse + un-shift + residual -> xmid (f32)
    wmma_gemm<EpiProjScatter><<<dim3(6, 512), 256, 0, stream>>>(
        opad, wproj, 384, 384, EpiProjScatter{xmid, x, proj_b});

    // LN2 -> f16
    ln2_to_f16<<<6272, 256, 0, stream>>>(xmid, n2g, n2b, h2);

    // fc1 GEMM + exact GELU: [50176,384] x [384,1536]
    wmma_gemm<EpiGeluHalf><<<dim3(24, 392), 256, 0, stream>>>(
        h2, wfc1, 384, 1536, EpiGeluHalf{h3, fc1_b, 1536});

    // fc2 GEMM + residual -> out: [50176,1536] x [1536,384]
    wmma_gemm<EpiFc2Out><<<dim3(6, 392), 256, 0, stream>>>(
        h3, wfc2, 1536, 384, EpiFc2Out{out, xmid, fc2_b, 384});
}
